// QuadraticHeterogenousCrossAttention_90563680403993
// MI455X (gfx1250) — matrix operand order repaired and checked by reading, not compiled, not run on hardware
//
#include <hip/hip_runtime.h>
#include <hip/hip_bf16.h>
#include <cstdint>
#include <cstddef>

typedef __bf16 v16bf __attribute__((ext_vector_type(16)));
typedef float  v8f   __attribute__((ext_vector_type(8)));

namespace {
constexpr int cB = 4, cH = 8, cDH = 64, cS = 1024, cD = 512;
constexpr float cY0 = 0.28209479177387814f;
constexpr float cY1 = 0.4886025119029199f;
}

union Frag {
  v16bf bf;
  uint4 u[2];
  unsigned short s[16];
};

__device__ __forceinline__ unsigned short f32_bf16(float f) {
  unsigned int u = __float_as_uint(f);
  u += 0x7FFFu + ((u >> 16) & 1u);           // round-to-nearest-even
  return (unsigned short)(u >> 16);
}
__device__ __forceinline__ float bf16_f32(unsigned short h) {
  return __uint_as_float((unsigned int)h << 16);
}
__device__ __forceinline__ uint4 zero4() {
  uint4 z; z.x = 0; z.y = 0; z.z = 0; z.w = 0; return z;
}
__device__ __forceinline__ float silu(float a) {
  return a / (1.0f + __expf(-a));
}

// ---------------------------------------------------------------------------
// One-shot f32 -> bf16 conversion (4 elements/thread, packed 8B stores)
// ---------------------------------------------------------------------------
__global__ void cvt_f32_bf16(const float* __restrict__ src,
                             unsigned short* __restrict__ dst, int n4) {
  int i = blockIdx.x * 256 + threadIdx.x;
  if (i >= n4) return;
  float4 v = ((const float4*)src)[i];
  union { unsigned short s[4]; uint2 u; } p;
  p.s[0] = f32_bf16(v.x);
  p.s[1] = f32_bf16(v.y);
  p.s[2] = f32_bf16(v.z);
  p.s[3] = f32_bf16(v.w);
  ((uint2*)dst)[i] = p.u;
}

// ---------------------------------------------------------------------------
// Weight transpose + convert: W f32 (K x N) -> Wt bf16 (N x K)
// ---------------------------------------------------------------------------
__global__ void wt_cvt(const float* __restrict__ W,
                       unsigned short* __restrict__ Wt, int N, int K) {
  int i = blockIdx.x * 256 + threadIdx.x;
  if (i >= N * K) return;
  int k = i / N, n = i % N;                   // coalesced read along n
  Wt[(size_t)n * K + k] = f32_bf16(W[i]);
}

// ---------------------------------------------------------------------------
// LDS-free bf16 WMMA GEMM: C[M,N] = A[M,K] @ Wt[N,K]^T + bias[N]  (f32 out)
// Fragments load directly from global (A row-major, Wt pre-transposed);
// reuse comes from L1/L2 (entire working set is L2-resident).
// block = 128 (4 waves); wave tile 16x64; block tile 64x64.
// ---------------------------------------------------------------------------
__global__ void gemm_bf16_direct(const unsigned short* __restrict__ A,
                                 const unsigned short* __restrict__ Wt,
                                 const float* __restrict__ bias,
                                 float* __restrict__ C,
                                 int N, int K) {
  const int lane = threadIdx.x & 31, wv = threadIdx.x >> 5;
  const int lg = lane >> 4, ln = lane & 15;
  const int m0 = blockIdx.y * 64, n0 = blockIdx.x * 64;

  const unsigned short* aRow  = A  + (size_t)(m0 + wv * 16 + ln) * K;
  const unsigned short* bRow0 = Wt + (size_t)(n0 + ln) * K + lg * 16;

  v8f acc[4];
#pragma unroll
  for (int i = 0; i < 4; ++i) acc[i] = {};

#pragma unroll 2
  for (int kc = 0; kc < K; kc += 32) {
    Frag aF;
    aF.u[0] = *(const uint4*)(aRow + kc + lg * 16);
    aF.u[1] = *(const uint4*)(aRow + kc + lg * 16 + 8);
#pragma unroll
    for (int nb = 0; nb < 4; ++nb) {
      Frag bF;
      const unsigned short* br = bRow0 + (size_t)(nb * 16) * K + kc;
      bF.u[0] = *(const uint4*)(br);
      bF.u[1] = *(const uint4*)(br + 8);
      acc[nb] = __builtin_amdgcn_wmma_f32_16x16x32_bf16(
          false, aF.bf, false, bF.bf, (short)0, acc[nb], false, false);
    }
  }

#pragma unroll
  for (int nb = 0; nb < 4; ++nb)
#pragma unroll
    for (int r = 0; r < 8; ++r) {
      int m = m0 + wv * 16 + r + 8 * lg;
      int n = n0 + nb * 16 + ln;
      C[(size_t)m * N + n] = acc[nb][r] + bias[n];
    }
}

// ---------------------------------------------------------------------------
// RoPE + layout conversion for Q: stage f32 (B,S,D) -> qb bf16 (B,H,S,DH)
// ---------------------------------------------------------------------------
__global__ void rope_q_kernel(const float* __restrict__ stage,
                              unsigned short* __restrict__ qb) {
  size_t idx = (size_t)blockIdx.x * blockDim.x + threadIdx.x;
  if (idx >= (size_t)cB * cS * cH * 32) return;
  int j = idx & 31; idx >>= 5;
  int h = idx & 7;  idx >>= 3;
  int s = (int)(idx & 1023);
  int b = (int)(idx >> 10);
  int pos = s >> 7;                                   // s / N, N=128
  float freq = __expf(-6.907755279f * (2.0f * j) / 64.0f);   // 1000^{-2j/DH}
  float sn, cs;
  __sincosf((float)pos * freq, &sn, &cs);
  const float* src = stage + ((size_t)b * cS + s) * cD + h * cDH;
  float t1 = src[2 * j], t2 = src[2 * j + 1];
  unsigned short* dst = qb + (((size_t)b * cH + h) * cS + s) * cDH;
  dst[2 * j]     = f32_bf16(t1 * cs - t2 * sn);
  dst[2 * j + 1] = f32_bf16(t1 * sn + t2 * cs);
}

// ---------------------------------------------------------------------------
// RoPE K + transpose V: stage f32 (B,S,2D) -> kbF bf16 (B,H,S,DH),
//                                             vtF bf16 (B,H,DH,S)
// ---------------------------------------------------------------------------
__global__ void rope_kv_kernel(const float* __restrict__ stage,
                               unsigned short* __restrict__ kbF,
                               unsigned short* __restrict__ vtF) {
  size_t idx = (size_t)blockIdx.x * blockDim.x + threadIdx.x;
  if (idx >= (size_t)cB * cS * cH * 32) return;
  int j = idx & 31; idx >>= 5;
  int h = idx & 7;  idx >>= 3;
  int s = (int)(idx & 1023);
  int b = (int)(idx >> 10);
  int pos = s >> 7;
  float freq = __expf(-6.907755279f * (2.0f * j) / 64.0f);
  float sn, cs;
  __sincosf((float)pos * freq, &sn, &cs);
  const float* row = stage + ((size_t)b * cS + s) * (2 * cD);
  float k1 = row[h * cDH + 2 * j], k2 = row[h * cDH + 2 * j + 1];
  unsigned short* kd = kbF + (((size_t)b * cH + h) * cS + s) * cDH;
  kd[2 * j]     = f32_bf16(k1 * cs - k2 * sn);
  kd[2 * j + 1] = f32_bf16(k1 * sn + k2 * cs);
  float v1 = row[cD + h * cDH + 2 * j], v2 = row[cD + h * cDH + 2 * j + 1];
  unsigned short* vbase = vtF + (((size_t)b * cH + h) * cDH) * cS;
  vbase[(size_t)(2 * j) * cS + s]     = f32_bf16(v1);
  vbase[(size_t)(2 * j + 1) * cS + s] = f32_bf16(v2);
}

// ---------------------------------------------------------------------------
// Spherical-harmonic bias MLP on the matrix pipe (3 WMMAs per 16 pairs).
// ---------------------------------------------------------------------------
__global__ void sh_bias_wmma_kernel(const float* __restrict__ x0,
                                    const float* __restrict__ W1, const float* __restrict__ b1,
                                    const float* __restrict__ W2, const float* __restrict__ b2,
                                    const float* __restrict__ W3, const float* __restrict__ b3,
                                    unsigned short* __restrict__ bias) {
  __shared__ unsigned short Ht[4][16 * 24];          // per-wave transpose slab
  const int lane = threadIdx.x & 31, wv = threadIdx.x >> 5;
  const int lg = lane >> 4, ln = lane & 15;

  const int bid = blockIdx.x;                        // B*S*(S/64) = 65536
  const int kb64 = bid & 15;
  const int qi = (bid >> 4) & 1023;
  const int b = bid >> 14;
  const int ki0 = kb64 * 64 + wv * 16;

  Frag w1f, w2f, w3f;
#pragma unroll
  for (int i = 0; i < 16; ++i) { w1f.s[i] = 0; w2f.s[i] = 0; w3f.s[i] = 0; }
  if (lg == 0) {                                     // K rows 0..15 live here
#pragma unroll
    for (int k = 0; k < 4; ++k)  w1f.s[k] = f32_bf16(W1[k * 16 + ln]);
#pragma unroll
    for (int k = 0; k < 16; ++k) w2f.s[k] = f32_bf16(W2[k * 16 + ln]);
    if (ln < 8) {
#pragma unroll
      for (int k = 0; k < 16; ++k) w3f.s[k] = f32_bf16(W3[k * 8 + ln]);
    }
  }
  const float b1v = b1[ln];
  const float b2v = b2[ln];
  const float b3v = (ln < 8) ? b3[ln] : 0.0f;

  Frag x;
#pragma unroll
  for (int i = 0; i < 16; ++i) x.s[i] = 0;
  if (lg == 0) {
    const float* cq = x0 + ((size_t)b * cS + qi) * cD;
    const float* ck = x0 + ((size_t)b * cS + ki0 + ln) * cD;
    float rx = cq[0] - ck[0], ry = cq[1] - ck[1], rz = cq[2] - ck[2];
    float inv = 1.0f / (sqrtf(rx * rx + ry * ry + rz * rz) + 1e-6f);
    x.s[0] = f32_bf16(cY0);
    x.s[1] = f32_bf16(cY1 * ry * inv);
    x.s[2] = f32_bf16(cY1 * rz * inv);
    x.s[3] = f32_bf16(cY1 * rx * inv);
  }

  v8f c1 = {};
  c1 = __builtin_amdgcn_wmma_f32_16x16x32_bf16(false, x.bf, false, w1f.bf,
                                               (short)0, c1, false, false);
#pragma unroll
  for (int r = 0; r < 8; ++r)
    Ht[wv][(r + 8 * lg) * 24 + ln] = f32_bf16(silu(c1[r] + b1v));
  Frag h1;
  h1.u[0] = lg ? zero4() : *(const uint4*)&Ht[wv][ln * 24];
  h1.u[1] = lg ? zero4() : *(const uint4*)&Ht[wv][ln * 24 + 8];

  v8f c2 = {};
  c2 = __builtin_amdgcn_wmma_f32_16x16x32_bf16(false, h1.bf, false, w2f.bf,
                                               (short)0, c2, false, false);
#pragma unroll
  for (int r = 0; r < 8; ++r)
    Ht[wv][(r + 8 * lg) * 24 + ln] = f32_bf16(silu(c2[r] + b2v));
  Frag h2;
  h2.u[0] = lg ? zero4() : *(const uint4*)&Ht[wv][ln * 24];
  h2.u[1] = lg ? zero4() : *(const uint4*)&Ht[wv][ln * 24 + 8];

  v8f c3 = {};
  c3 = __builtin_amdgcn_wmma_f32_16x16x32_bf16(false, h2.bf, false, w3f.bf,
                                               (short)0, c3, false, false);
  if (ln < 8) {                                       // N = head = ln
    unsigned short* dst =
        bias + (((size_t)b * cH + ln) * cS + qi) * cS + ki0 + 8 * lg;
#pragma unroll
    for (int r = 0; r < 8; ++r) dst[r] = f32_bf16(c3[r] + b3v);
  }
}

// ---------------------------------------------------------------------------
// Fused flash attention. Output written DIRECTLY as bf16 (B,S,D) so the
// final projection GEMM consumes it with no staging pass.
// ---------------------------------------------------------------------------
__global__ void attn_kernel(const unsigned short* __restrict__ qb,
                            const unsigned short* __restrict__ kb,
                            const unsigned short* __restrict__ vt,
                            const unsigned short* __restrict__ bias,
                            const float* __restrict__ fw,
                            const float* __restrict__ denom,
                            unsigned short* __restrict__ out) {
  __shared__ unsigned short Pt[4][16 * 40];     // per-wave P transpose slab
  const int lane = threadIdx.x & 31, wv = threadIdx.x >> 5;
  const int lg = lane >> 4, ln = lane & 15;
  const int b = blockIdx.z, h = blockIdx.y;
  const int q0 = blockIdx.x * 64 + wv * 16;

  float w0 = fw[0], w1 = fw[1], w2 = fw[2];
  float wm = fmaxf(w0, fmaxf(w1, w2));
  float e0 = __expf(w0 - wm), e1 = __expf(w1 - wm), e2 = __expf(w2 - wm);
  float ei = 1.0f / (e0 + e1 + e2);
  float gate[3] = {e0 * ei, e1 * ei, e2 * ei};
  const float invd = 1.0f / denom[h];

  Frag qA[2];
  const unsigned short* qrow = qb + (((size_t)b * cH + h) * cS + (q0 + ln)) * cDH;
#pragma unroll
  for (int kc = 0; kc < 2; ++kc) {
    qA[kc].u[0] = *(const uint4*)(qrow + kc * 32 + lg * 16);
    qA[kc].u[1] = *(const uint4*)(qrow + kc * 32 + lg * 16 + 8);
  }

  v8f outAcc[4];
#pragma unroll
  for (int i = 0; i < 4; ++i) outAcc[i] = {};
  const unsigned short* biasBase = bias + ((size_t)b * cH + h) * cS * cS;

  for (int f = 0; f < 3; ++f) {
    const unsigned short* kbf = kb + (((size_t)f * cB + b) * cH + h) * (size_t)cS * cDH;
    const unsigned short* vtf = vt + (((size_t)f * cB + b) * cH + h) * (size_t)cDH * cS;

    float mrow[8], lrow[8];
    v8f O[4];
#pragma unroll
    for (int r = 0; r < 8; ++r) { mrow[r] = -3.0e38f; lrow[r] = 0.0f; }
#pragma unroll
    for (int i = 0; i < 4; ++i) O[i] = {};

    for (int k0 = 0; k0 < cS; k0 += 32) {
      v8f sc[2] = {v8f{}, v8f{}};
#pragma unroll
      for (int kt = 0; kt < 2; ++kt) {
        const unsigned short* krow = kbf + (size_t)(k0 + kt * 16 + ln) * cDH;
#pragma unroll
        for (int kc = 0; kc < 2; ++kc) {
          Frag kF;
          kF.u[0] = *(const uint4*)(krow + kc * 32 + lg * 16);
          kF.u[1] = *(const uint4*)(krow + kc * 32 + lg * 16 + 8);
          sc[kt] = __builtin_amdgcn_wmma_f32_16x16x32_bf16(
              false, qA[kc].bf, false, kF.bf, (short)0, sc[kt], false, false);
        }
      }
#pragma unroll
      for (int r = 0; r < 8; ++r) {
        int qi = q0 + r + 8 * lg;
        float s0 = sc[0][r] * invd +
                   bf16_f32(biasBase[(size_t)qi * cS + k0 + ln]);
        float s1 = sc[1][r] * invd +
                   bf16_f32(biasBase[(size_t)qi * cS + k0 + 16 + ln]);
        float mr = fmaxf(s0, s1);
#pragma unroll
        for (int off = 1; off < 16; off <<= 1)
          mr = fmaxf(mr, __shfl_xor(mr, off, 32));
        float newm = fmaxf(mrow[r], mr);
        float alpha = __expf(mrow[r] - newm);
        mrow[r] = newm;
        float p0 = __expf(s0 - newm);
        float p1 = __expf(s1 - newm);
        float rs = p0 + p1;
#pragma unroll
        for (int off = 1; off < 16; off <<= 1)
          rs += __shfl_xor(rs, off, 32);
        lrow[r] = lrow[r] * alpha + rs;
#pragma unroll
        for (int nb = 0; nb < 4; ++nb) O[nb][r] *= alpha;
        Pt[wv][(r + 8 * lg) * 40 + ln]      = f32_bf16(p0);
        Pt[wv][(r + 8 * lg) * 40 + 16 + ln] = f32_bf16(p1);
      }
      Frag pA;
      pA.u[0] = *(const uint4*)&Pt[wv][ln * 40 + lg * 16];
      pA.u[1] = *(const uint4*)&Pt[wv][ln * 40 + lg * 16 + 8];
#pragma unroll
      for (int nb = 0; nb < 4; ++nb) {
        Frag vF;
        const unsigned short* vrow = vtf + (size_t)(nb * 16 + ln) * cS + k0 + lg * 16;
        vF.u[0] = *(const uint4*)(vrow);
        vF.u[1] = *(const uint4*)(vrow + 8);
        O[nb] = __builtin_amdgcn_wmma_f32_16x16x32_bf16(
            false, pA.bf, false, vF.bf, (short)0, O[nb], false, false);
      }
    }
#pragma unroll
    for (int nb = 0; nb < 4; ++nb)
#pragma unroll
      for (int r = 0; r < 8; ++r)
        outAcc[nb][r] += gate[f] * O[nb][r] / lrow[r];
  }

#pragma unroll
  for (int nb = 0; nb < 4; ++nb)
#pragma unroll
    for (int r = 0; r < 8; ++r) {
      int qi = q0 + r + 8 * lg;
      out[((size_t)b * cS + qi) * cD + h * cDH + nb * 16 + ln] =
          f32_bf16(outAcc[nb][r]);
    }
}

// ---------------------------------------------------------------------------
extern "C" void kernel_launch(void* const* d_in, const int* in_sizes, int n_in,
                              void* d_out, int out_size, void* d_ws, size_t ws_size,
                              hipStream_t stream) {
  (void)in_sizes; (void)n_in; (void)out_size; (void)ws_size;
  const float* x0  = (const float*)d_in[0];
  const float* v0  = (const float*)d_in[1];
  const float* cf  = (const float*)d_in[2];
  const float* qd  = (const float*)d_in[3];
  const float* Wq  = (const float*)d_in[4];
  const float* bq  = (const float*)d_in[5];
  const float* Wkv = (const float*)d_in[6];
  const float* bkv = (const float*)d_in[7];
  const float* Wo  = (const float*)d_in[8];
  const float* bo  = (const float*)d_in[9];
  const float* fw  = (const float*)d_in[10];
  const float* ad  = (const float*)d_in[11];
  const float* W1  = (const float*)d_in[12];
  const float* b1  = (const float*)d_in[13];
  const float* W2  = (const float*)d_in[14];
  const float* b2  = (const float*)d_in[15];
  const float* W3  = (const float*)d_in[16];
  const float* b3  = (const float*)d_in[17];
  float* out = (float*)d_out;

  char* ws = (char*)d_ws;
  float*          stage = (float*)ws;                              // 16 MiB
  unsigned short* qb    = (unsigned short*)(ws + (size_t)16777216);
  unsigned short* kb    = (unsigned short*)(ws + (size_t)20971520);
  unsigned short* vt    = (unsigned short*)(ws + (size_t)33554432);
  unsigned short* bias  = (unsigned short*)(ws + (size_t)46137344);
  unsigned short* actb  = (unsigned short*)(ws + (size_t)113246208); // 4 MiB bf16 act
  unsigned short* wtb   = (unsigned short*)(ws + (size_t)117440512); // 1 MiB bf16 Wt

  const size_t perF = (size_t)cB * cH * cS * cDH;
  const int M = cB * cS;                    // 4096
  const int actN4 = M * cD / 4;             // f32->bf16 granules for activations

  // 1) SH bias on the matrix pipe
  sh_bias_wmma_kernel<<<dim3(65536), dim3(128), 0, stream>>>(
      x0, W1, b1, W2, b2, W3, b3, bias);

  // 2) Q projection + RoPE  (operands pre-converted to bf16 once)
  cvt_f32_bf16<<<dim3((actN4 + 255) / 256), dim3(256), 0, stream>>>(qd, actb, actN4);
  wt_cvt<<<dim3(512 * 512 / 256), dim3(256), 0, stream>>>(Wq, wtb, 512, 512);
  gemm_bf16_direct<<<dim3(8, 64), dim3(128), 0, stream>>>(actb, wtb, bq, stage, 512, 512);
  rope_q_kernel<<<dim3(4096), dim3(256), 0, stream>>>(stage, qb);

  // 3) KV projections per feature (actb/wtb reused; stream-serialized)
  const float* feats[3] = {x0, v0, cf};
  for (int f = 0; f < 3; ++f) {
    cvt_f32_bf16<<<dim3((actN4 + 255) / 256), dim3(256), 0, stream>>>(feats[f], actb, actN4);
    wt_cvt<<<dim3(1024 * 512 / 256), dim3(256), 0, stream>>>(
        Wkv + (size_t)f * 512 * 1024, wtb, 1024, 512);
    gemm_bf16_direct<<<dim3(16, 64), dim3(128), 0, stream>>>(
        actb, wtb, bkv + (size_t)f * 1024, stage, 1024, 512);
    rope_kv_kernel<<<dim3(4096), dim3(256), 0, stream>>>(
        stage, kb + (size_t)f * perF, vt + (size_t)f * perF);
  }

  // 4) fused flash attention; writes bf16 activation for the final GEMM
  attn_kernel<<<dim3(16, 8, 4), dim3(128), 0, stream>>>(qb, kb, vt, bias, fw, ad, actb);

  // 5) output projection
  wt_cvt<<<dim3(512 * 512 / 256), dim3(256), 0, stream>>>(Wo, wtb, 512, 512);
  gemm_bf16_direct<<<dim3(8, 64), dim3(128), 0, stream>>>(actb, wtb, bo, out, 512, 512);
}
